// DeformableTransformer_54992761258336
// MI455X (gfx1250) — compile-verified, matrix operand order checked by reading, not run on hardware
//
#include <hip/hip_runtime.h>
#include <cstdint>
#include <cstddef>

// ---------------- problem constants ----------------
#define DM     256
#define HEADS  8
#define HD     32
#define LVLS   4
#define PTS    4
#define TFRM   4
#define TALL   6
#define NQv    100
#define BSv    2
#define SPT    1360
#define NETOK  (BSv*TFRM*SPT)   // 10880
#define NDTOK  (BSv*TALL*NQv)   // 1200
#define FFN    1024
#define WPACK_TOTAL 12386304    // sum of K*Mo over all packed weights

typedef __attribute__((ext_vector_type(16))) _Float16 v16h;
typedef __attribute__((ext_vector_type(8)))  float    v8f;

__device__ const int c_lsi[4]  = {0, 1024, 1280, 1344};
__device__ const int c_wdim[4] = {32, 16, 8, 4};

// ---------------- device scratch (module-allocated; graph-capture safe) ----------------
__device__ __align__(16) float g_x   [NETOK*DM];
__device__ __align__(16) float g_pos [NETOK*DM];
__device__ __align__(16) float g_q   [NETOK*DM];
__device__ __align__(16) float g_val [NETOK*DM];
__device__ __align__(16) float g_off [NETOK*DM];
__device__ __align__(16) float g_aw  [NETOK*128];
__device__ __align__(16) float g_attn[NETOK*DM];
__device__ __align__(16) float g_tmp [NETOK*DM];
__device__ __align__(16) float g_ffn [NETOK*FFN];

__device__ __align__(16) float g_xd   [NDTOK*DM];
__device__ __align__(16) float g_qpv  [NDTOK*DM];
__device__ __align__(16) float g_qd   [NDTOK*DM];
__device__ __align__(16) float g_qkv  [NDTOK*768];
__device__ __align__(16) float g_sa   [NDTOK*DM];
__device__ __align__(16) float g_offd [NDTOK*1024];
__device__ __align__(16) float g_awd  [NDTOK*512];
__device__ __align__(16) float g_attnd[NDTOK*DM];
__device__ __align__(16) float g_tmpd [NDTOK*DM];
__device__ __align__(16) float g_ffnd [NDTOK*FFN];
__device__ __align__(16) float g_refp [NDTOK*2];

__device__ __align__(32) _Float16 g_wpack[WPACK_TOTAL];

// ---------------- weight repack: f32 row-major (K x Mo) -> per-lane WMMA B fragments ----
// layout: Wp[((mb*KB + kb)*32 + lane)*16 + j] = W[k, mb*16 + (lane&15)]
//   k = kb*32 + ((j<8)?j:j+8) + ((lane&16)?8:0)   (mirrors the 16-bit A/B fragment map)
__global__ void k_pack(const float* __restrict__ W, _Float16* __restrict__ dst, int K, int Mo) {
  int idx = blockIdx.x * blockDim.x + threadIdx.x;
  int total = K * Mo;
  if (idx >= total) return;
  int j    = idx & 15;
  int lane = (idx >> 4) & 31;
  int blk  = idx >> 9;
  int KB   = K >> 5;
  int kb   = blk % KB;
  int mb   = blk / KB;
  int col  = (mb << 4) + (lane & 15);
  int k    = (kb << 5) + ((j < 8) ? j : j + 8) + ((lane & 16) ? 8 : 0);
  dst[idx] = (_Float16)W[(size_t)k * Mo + col];
}

// A-fragment loader: 16 f32 (two 32B runs) -> v16h, per the ISA 16-bit A layout
__device__ __forceinline__ v16h load_afrag(const float* __restrict__ ap) {
  float4 a0 = *(const float4*)(ap + 0);
  float4 a1 = *(const float4*)(ap + 4);
  float4 a2 = *(const float4*)(ap + 16);
  float4 a3 = *(const float4*)(ap + 20);
  v16h af;
  af[0]=(_Float16)a0.x;  af[1]=(_Float16)a0.y;  af[2]=(_Float16)a0.z;  af[3]=(_Float16)a0.w;
  af[4]=(_Float16)a1.x;  af[5]=(_Float16)a1.y;  af[6]=(_Float16)a1.z;  af[7]=(_Float16)a1.w;
  af[8]=(_Float16)a2.x;  af[9]=(_Float16)a2.y;  af[10]=(_Float16)a2.z; af[11]=(_Float16)a2.w;
  af[12]=(_Float16)a3.x; af[13]=(_Float16)a3.y; af[14]=(_Float16)a3.z; af[15]=(_Float16)a3.w;
  return af;
}

// ---------------- WMMA GEMM: C[r, c_col_off + col] = act(A[r,:K] @ W + bias) -----------
// one wave -> 32 rows x 64 cols (2 A-frags x 4 B-frags -> 8 WMMA per K-step of 32)
// block = 8 waves (256 rows); grid.y covers mo_count/64
__global__ void __launch_bounds__(256)
k_gemm(const float* __restrict__ A, const _Float16* __restrict__ Wp,
       const float* __restrict__ bias, float* __restrict__ C,
       int N, int K, int mb_start, int c_stride, int c_col_off, int act) {
  const int KB   = K >> 5;
  const int lane = threadIdx.x & 31;
  const int wave = threadIdx.x >> 5;
  const int rowTile = blockIdx.x * 8 + wave;
  const int row0 = rowTile << 5;             // 32 rows per wave tile
  if (row0 >= N) return;                     // wave-uniform exit: EXEC stays all-ones for WMMA
  const int lrow = lane & 15;
  const int hi   = (lane & 16) ? 1 : 0;
  int ar0 = row0 + lrow;       if (ar0 >= N) ar0 = N - 1;
  int ar1 = row0 + 16 + lrow;  if (ar1 >= N) ar1 = N - 1;
  const float* __restrict__ Arow0 = A + (size_t)ar0 * K + (hi ? 8 : 0);
  const float* __restrict__ Arow1 = A + (size_t)ar1 * K + (hi ? 8 : 0);
  const int mbBase = mb_start + blockIdx.y * 4;
  const _Float16* __restrict__ W0 = Wp + ((size_t)(mbBase + 0) * KB) * 512 + lane * 16;
  const _Float16* __restrict__ W1 = Wp + ((size_t)(mbBase + 1) * KB) * 512 + lane * 16;
  const _Float16* __restrict__ W2 = Wp + ((size_t)(mbBase + 2) * KB) * 512 + lane * 16;
  const _Float16* __restrict__ W3 = Wp + ((size_t)(mbBase + 3) * KB) * 512 + lane * 16;
  v8f a00 = {}, a01 = {}, a02 = {}, a03 = {};
  v8f a10 = {}, a11 = {}, a12 = {}, a13 = {};
  for (int kb = 0; kb < KB; ++kb) {
    const float* ap0 = Arow0 + (kb << 5);
    const float* ap1 = Arow1 + (kb << 5);
    if (kb + 1 < KB) {                        // WGP-scope prefetch: fill all cache levels
      __builtin_prefetch(ap0 + 32, 0, 3);     // global_prefetch_b8
      __builtin_prefetch(ap1 + 32, 0, 3);
    }
    v16h af0 = load_afrag(ap0);
    v16h af1 = load_afrag(ap1);
    const size_t ko = (size_t)kb * 512;
    v16h b0 = *(const v16h*)(W0 + ko);
    v16h b1 = *(const v16h*)(W1 + ko);
    v16h b2 = *(const v16h*)(W2 + ko);
    v16h b3 = *(const v16h*)(W3 + ko);
    a00 = __builtin_amdgcn_wmma_f32_16x16x32_f16(false, af0, false, b0, (short)0, a00, false, false);
    a01 = __builtin_amdgcn_wmma_f32_16x16x32_f16(false, af0, false, b1, (short)0, a01, false, false);
    a02 = __builtin_amdgcn_wmma_f32_16x16x32_f16(false, af0, false, b2, (short)0, a02, false, false);
    a03 = __builtin_amdgcn_wmma_f32_16x16x32_f16(false, af0, false, b3, (short)0, a03, false, false);
    a10 = __builtin_amdgcn_wmma_f32_16x16x32_f16(false, af1, false, b0, (short)0, a10, false, false);
    a11 = __builtin_amdgcn_wmma_f32_16x16x32_f16(false, af1, false, b1, (short)0, a11, false, false);
    a12 = __builtin_amdgcn_wmma_f32_16x16x32_f16(false, af1, false, b2, (short)0, a12, false, false);
    a13 = __builtin_amdgcn_wmma_f32_16x16x32_f16(false, af1, false, b3, (short)0, a13, false, false);
  }
  const int coln  = lane & 15;
  const int ccol0 = c_col_off + blockIdx.y * 64 + coln;
#define GEMM_EPILOG(ACCV, CB, RB)                                          \
  {                                                                        \
    float bv = bias ? bias[(mbBase + (CB)) * 16 + coln] : 0.0f;            \
    int rb = row0 + (RB)*16 + (hi ? 8 : 0);                                \
    _Pragma("unroll")                                                      \
    for (int i = 0; i < 8; ++i) {                                          \
      int r = rb + i;                                                      \
      if (r < N) {                                                         \
        float v = ACCV[i] + bv;                                            \
        if (act) v = fmaxf(v, 0.0f);                                       \
        C[(size_t)r * c_stride + ccol0 + ((CB) << 4)] = v;                 \
      }                                                                    \
    }                                                                      \
  }
  GEMM_EPILOG(a00, 0, 0) GEMM_EPILOG(a01, 1, 0) GEMM_EPILOG(a02, 2, 0) GEMM_EPILOG(a03, 3, 0)
  GEMM_EPILOG(a10, 0, 1) GEMM_EPILOG(a11, 1, 1) GEMM_EPILOG(a12, 2, 1) GEMM_EPILOG(a13, 3, 1)
#undef GEMM_EPILOG
}

// ---------------- fused residual + LayerNorm (one block per row, D=256) ----------------
__global__ void __launch_bounds__(256)
k_ln(float* __restrict__ out, const float* __restrict__ a, const float* __restrict__ res,
     const float* __restrict__ g, const float* __restrict__ b) {
  __shared__ float s1[256], s2[256];
  int row = blockIdx.x;
  int t = threadIdx.x;
  float v = a[(size_t)row * DM + t];
  if (res) v += res[(size_t)row * DM + t];
  s1[t] = v; s2[t] = v * v;
  __syncthreads();
  for (int off = 128; off > 0; off >>= 1) {
    if (t < off) { s1[t] += s1[t + off]; s2[t] += s2[t + off]; }
    __syncthreads();
  }
  float mu  = s1[0] * (1.0f / DM);
  float var = s2[0] * (1.0f / DM) - mu * mu;
  float inv = rsqrtf(var + 1e-5f);
  out[(size_t)row * DM + t] = (v - mu) * inv * g[t] + b[t];
}

// ---------------- elementwise helpers ----------------
__global__ void k_add(float* __restrict__ o, const float* __restrict__ a,
                      const float* __restrict__ b, int n) {
  int i = blockIdx.x * blockDim.x + threadIdx.x;
  if (i < n) o[i] = a[i] + b[i];
}
__global__ void k_copy(float* __restrict__ o, const float* __restrict__ a, int n) {
  int i = blockIdx.x * blockDim.x + threadIdx.x;
  if (i < n) o[i] = a[i];
}

// (B,C,T,H,W) -> (B,T,SPT,DM) at spatial offset sOff; optional +level_embed
__global__ void k_flatten(const float* __restrict__ src, const float* __restrict__ lvl,
                          float* __restrict__ dst, int Hs, int Ws, int sOff, int total) {
  int idx = blockIdx.x * blockDim.x + threadIdx.x;
  if (idx >= total) return;
  int c = idx % DM;
  int r = idx / DM;
  int x = r % Ws; r /= Ws;
  int y = r % Hs; r /= Hs;
  int t = r % TFRM; int b2 = r / TFRM;
  float v = src[((((size_t)b2 * DM + c) * TFRM + t) * Hs + y) * Ws + x];
  if (lvl) v += lvl[c];
  dst[(((size_t)(b2 * TFRM + t) * SPT) + sOff + (size_t)y * Ws + x) * DM + c] = v;
}

// softmax over trailing L (rows = Ntok*HEADS, contiguous per head)
__global__ void k_softmax(float* __restrict__ a, int rows, int L) {
  int r = blockIdx.x * blockDim.x + threadIdx.x;
  if (r >= rows) return;
  float* p = a + (size_t)r * L;
  float m = -1e30f;
  for (int i = 0; i < L; ++i) m = fmaxf(m, p[i]);
  float s = 0.f;
  for (int i = 0; i < L; ++i) s += __expf(p[i] - m);
  float inv = 1.0f / s;
  for (int i = 0; i < L; ++i) p[i] = __expf(p[i] - m) * inv;
}

// ---------------- MSDA bilinear samplers (valid_ratios == 1, masks all False) ----------
__global__ void k_msda_enc(const float* __restrict__ val, const float* __restrict__ off,
                           const float* __restrict__ aw, float* __restrict__ out) {
  int gid = blockIdx.x * blockDim.x + threadIdx.x;
  if (gid >= NETOK * HEADS) return;
  int h = gid & 7;
  int n = gid >> 3;
  int s = n % SPT;
  int bt = n / SPT;
  int lvl, rem;
  if      (s < 1024) { lvl = 0; rem = s; }
  else if (s < 1280) { lvl = 1; rem = s - 1024; }
  else if (s < 1344) { lvl = 2; rem = s - 1280; }
  else               { lvl = 3; rem = s - 1344; }
  int wq = c_wdim[lvl];
  float rx = ((rem % wq) + 0.5f) / wq;
  float ry = ((rem / wq) + 0.5f) / wq;
  const float* op = off + (size_t)n * (HEADS*LVLS*PTS*2) + h * (LVLS*PTS*2);
  const float* ap = aw  + (size_t)n * (HEADS*LVLS*PTS)   + h * (LVLS*PTS);
  float acc[HD];
#pragma unroll
  for (int d = 0; d < HD; ++d) acc[d] = 0.f;
  for (int l = 0; l < LVLS; ++l) {
    int Hs = c_wdim[l];
    const float* vb = val + ((size_t)bt * SPT + c_lsi[l]) * DM + h * HD;
    for (int p = 0; p < PTS; ++p) {
      int si = l * PTS + p;
      float a = ap[si];
      float x = (rx + op[si*2+0] / Hs) * Hs - 0.5f;
      float y = (ry + op[si*2+1] / Hs) * Hs - 0.5f;
      float xf = floorf(x), yf = floorf(y);
      float dx = x - xf, dy = y - yf;
      int x0 = (int)xf, y0 = (int)yf;
#pragma unroll
      for (int tap = 0; tap < 4; ++tap) {
        int xx = x0 + (tap & 1);
        int yy = y0 + (tap >> 1);
        if (xx >= 0 && xx < Hs && yy >= 0 && yy < Hs) {
          float w = ((tap & 1) ? dx : 1.f - dx) * ((tap >> 1) ? dy : 1.f - dy) * a;
          const float* vp = vb + (size_t)(yy * Hs + xx) * DM;
#pragma unroll
          for (int d = 0; d < HD; ++d) acc[d] += w * vp[d];
        }
      }
    }
  }
  float* o2 = out + (size_t)n * DM + h * HD;
#pragma unroll
  for (int d = 0; d < HD; ++d) o2[d] = acc[d];
}

__global__ void k_msda_dec(const float* __restrict__ val, const float* __restrict__ off,
                           const float* __restrict__ aw, const float* __restrict__ refp,
                           float* __restrict__ out) {
  int gid = blockIdx.x * blockDim.x + threadIdx.x;
  if (gid >= NDTOK * HEADS) return;
  int h = gid & 7;
  int n = gid >> 3;
  int b2 = n / (TALL * NQv);
  float rx = refp[(size_t)n * 2 + 0];
  float ry = refp[(size_t)n * 2 + 1];
  const float* op = off + (size_t)n * 1024 + h * 128;  // [t][l][p][2]
  const float* ap = aw  + (size_t)n * 512  + h * 64;   // [t][l][p]
  float acc[HD];
#pragma unroll
  for (int d = 0; d < HD; ++d) acc[d] = 0.f;
  for (int t = 0; t < TFRM; ++t) {
    for (int l = 0; l < LVLS; ++l) {
      int Hs = c_wdim[l];
      const float* vb = val + ((size_t)(b2 * TFRM + t) * SPT + c_lsi[l]) * DM + h * HD;
      for (int p = 0; p < PTS; ++p) {
        int si = (t * LVLS + l) * PTS + p;
        float a = ap[si];
        float x = (rx + op[si*2+0] / Hs) * Hs - 0.5f;
        float y = (ry + op[si*2+1] / Hs) * Hs - 0.5f;
        float xf = floorf(x), yf = floorf(y);
        float dx = x - xf, dy = y - yf;
        int x0 = (int)xf, y0 = (int)yf;
#pragma unroll
        for (int tap = 0; tap < 4; ++tap) {
          int xx = x0 + (tap & 1);
          int yy = y0 + (tap >> 1);
          if (xx >= 0 && xx < Hs && yy >= 0 && yy < Hs) {
            float w = ((tap & 1) ? dx : 1.f - dx) * ((tap >> 1) ? dy : 1.f - dy) * a;
            const float* vp = vb + (size_t)(yy * Hs + xx) * DM;
#pragma unroll
            for (int d = 0; d < HD; ++d) acc[d] += w * vp[d];
          }
        }
      }
    }
  }
  float* o2 = out + (size_t)n * DM + h * HD;
#pragma unroll
  for (int d = 0; d < HD; ++d) o2[d] = acc[d];
}

// ---------------- decoder self-attention (online softmax over 600 keys) ---------------
__global__ void k_mha(const float* __restrict__ qkv, float* __restrict__ out) {
  int gid = blockIdx.x * blockDim.x + threadIdx.x;
  const int NKV = TALL * NQv;  // 600
  if (gid >= BSv * HEADS * NKV) return;
  int qi = gid % NKV;
  int h  = (gid / NKV) & 7;
  int b2 = gid / (NKV * HEADS);
  const float* qp = qkv + (size_t)(b2 * NKV + qi) * 768 + h * HD;
  float qr[HD];
#pragma unroll
  for (int d = 0; d < HD; ++d) qr[d] = qp[d] * 0.17677669529663687f;  // 1/sqrt(32)
  float m = -1e30f, ssum = 0.f;
  float acc[HD];
#pragma unroll
  for (int d = 0; d < HD; ++d) acc[d] = 0.f;
  for (int k = 0; k < NKV; ++k) {
    const float* kp = qkv + (size_t)(b2 * NKV + k) * 768 + 256 + h * HD;
    float s = 0.f;
#pragma unroll
    for (int d = 0; d < HD; ++d) s += qr[d] * kp[d];
    float nm = fmaxf(m, s);
    float cf = __expf(m - nm);
    float pf = __expf(s - nm);
    ssum = ssum * cf + pf;
    m = nm;
    const float* vp = kp + 256;
#pragma unroll
    for (int d = 0; d < HD; ++d) acc[d] = acc[d] * cf + pf * vp[d];
  }
  float inv = 1.0f / ssum;
  float* o2 = out + (size_t)(b2 * NKV + qi) * DM + h * HD;
#pragma unroll
  for (int d = 0; d < HD; ++d) o2[d] = acc[d] * inv;
}

// ---------------- decoder init: qp = qe[:, :256] + temb[T]; tgt = qe[:, 256:] ----------
__global__ void k_decinit(const float* __restrict__ qe, const float* __restrict__ temb,
                          float* __restrict__ qp, float* __restrict__ tgt) {
  int idx = blockIdx.x * blockDim.x + threadIdx.x;
  if (idx >= NDTOK * DM) return;
  int c = idx % DM;
  int q = (idx / DM) % NQv;
  int T = (idx / (DM * NQv)) % TALL;
  qp[idx]  = qe[(size_t)q * (2 * DM) + c] + temb[(size_t)T * DM + c];
  tgt[idx] = qe[(size_t)q * (2 * DM) + DM + c];
}

// refp = sigmoid(qp @ W(256x2) + b)
__global__ void k_refp(const float* __restrict__ qp, const float* __restrict__ w,
                       const float* __restrict__ b, float* __restrict__ refp) {
  int n = blockIdx.x * blockDim.x + threadIdx.x;
  if (n >= NDTOK) return;
  const float* x = qp + (size_t)n * DM;
  float s0 = b[0], s1 = b[1];
  for (int c = 0; c < DM; ++c) { float v = x[c]; s0 += v * w[c*2+0]; s1 += v * w[c*2+1]; }
  refp[(size_t)n*2+0] = 1.0f / (1.0f + __expf(-s0));
  refp[(size_t)n*2+1] = 1.0f / (1.0f + __expf(-s1));
}

// ---------------- host orchestration ----------------
extern "C" void kernel_launch(void* const* d_in, const int* in_sizes, int n_in,
                              void* d_out, int out_size, void* d_ws, size_t ws_size,
                              hipStream_t stream) {
  (void)in_sizes; (void)out_size; (void)d_ws; (void)ws_size;
  if (n_in < 245) return;  // expects 13 tensors + 232 param leaves (insertion-order flatten)

  float *x, *pos, *q, *val, *off, *aw, *attn, *tmp, *ffn;
  float *xd, *qpv, *qd, *qkv, *sa, *offd, *awd, *attnd, *tmpd, *ffnd, *refp;
  _Float16* wpk;
  hipGetSymbolAddress((void**)&x,    HIP_SYMBOL(g_x));
  hipGetSymbolAddress((void**)&pos,  HIP_SYMBOL(g_pos));
  hipGetSymbolAddress((void**)&q,    HIP_SYMBOL(g_q));
  hipGetSymbolAddress((void**)&val,  HIP_SYMBOL(g_val));
  hipGetSymbolAddress((void**)&off,  HIP_SYMBOL(g_off));
  hipGetSymbolAddress((void**)&aw,   HIP_SYMBOL(g_aw));
  hipGetSymbolAddress((void**)&attn, HIP_SYMBOL(g_attn));
  hipGetSymbolAddress((void**)&tmp,  HIP_SYMBOL(g_tmp));
  hipGetSymbolAddress((void**)&ffn,  HIP_SYMBOL(g_ffn));
  hipGetSymbolAddress((void**)&xd,   HIP_SYMBOL(g_xd));
  hipGetSymbolAddress((void**)&qpv,  HIP_SYMBOL(g_qpv));
  hipGetSymbolAddress((void**)&qd,   HIP_SYMBOL(g_qd));
  hipGetSymbolAddress((void**)&qkv,  HIP_SYMBOL(g_qkv));
  hipGetSymbolAddress((void**)&sa,   HIP_SYMBOL(g_sa));
  hipGetSymbolAddress((void**)&offd, HIP_SYMBOL(g_offd));
  hipGetSymbolAddress((void**)&awd,  HIP_SYMBOL(g_awd));
  hipGetSymbolAddress((void**)&attnd,HIP_SYMBOL(g_attnd));
  hipGetSymbolAddress((void**)&tmpd, HIP_SYMBOL(g_tmpd));
  hipGetSymbolAddress((void**)&ffnd, HIP_SYMBOL(g_ffnd));
  hipGetSymbolAddress((void**)&refp, HIP_SYMBOL(g_refp));
  hipGetSymbolAddress((void**)&wpk,  HIP_SYMBOL(g_wpack));

  auto F = [&](int i) -> const float* { return (const float*)d_in[i]; };

  // ---- repack all GEMM weights to WMMA fragment layout ----
  size_t woff = 0;
  auto pack = [&](int idx, int K, int Mo) -> size_t {
    size_t o = woff;
    int total = K * Mo;
    k_pack<<<(total + 255) / 256, 256, 0, stream>>>(F(idx), wpk + o, K, Mo);
    woff += (size_t)total;
    return o;
  };
  size_t e_offw[6], e_aww[6], e_valw[6], e_outw[6], e_l1w[6], e_l2w[6];
  for (int i = 0; i < 6; ++i) {
    int B0 = 13 + 16 * i;   // off.w/b, aw.w/b, val.w/b, out.w/b, n1.g/b, l1.w/b, l2.w/b, n2.g/b
    e_offw[i] = pack(B0 + 0,  DM, 256);
    e_aww[i]  = pack(B0 + 2,  DM, 128);
    e_valw[i] = pack(B0 + 4,  DM, 256);
    e_outw[i] = pack(B0 + 6,  DM, 256);
    e_l1w[i]  = pack(B0 + 10, DM, FFN);
    e_l2w[i]  = pack(B0 + 12, FFN, DM);
  }
  size_t d_offw[6], d_aww[6], d_valw[6], d_outw[6], d_winw[6], d_woutw[6], d_l1w[6], d_l2w[6];
  for (int i = 0; i < 6; ++i) {
    int B0 = 13 + 96 + 22 * i;  // msda(8), n1(2), win, bin, wout(2), n2(2), l1(2), l2(2), n3(2)
    d_offw[i]  = pack(B0 + 0,  DM, 1024);
    d_aww[i]   = pack(B0 + 2,  DM, 512);
    d_valw[i]  = pack(B0 + 4,  DM, 256);
    d_outw[i]  = pack(B0 + 6,  DM, 256);
    d_winw[i]  = pack(B0 + 10, DM, 768);
    d_woutw[i] = pack(B0 + 12, DM, 256);
    d_l1w[i]   = pack(B0 + 16, DM, FFN);
    d_l2w[i]   = pack(B0 + 18, FFN, DM);
  }

  auto gemm = [&](const float* A, size_t wo, const float* bias, float* C,
                  int N, int K, int mb_start, int mo_count, int c_stride, int c_col_off, int act) {
    dim3 grid((N + 255) / 256, mo_count / 64);
    k_gemm<<<grid, 256, 0, stream>>>(A, wpk + wo, bias, C, N, K, mb_start, c_stride, c_col_off, act);
  };

  // ---- flatten sources and positional embeds (+level_embed) ----
  const float* lvl_embed = F(241);
  const int hw[4] = {32, 16, 8, 4};
  int sOff = 0;
  for (int l = 0; l < 4; ++l) {
    int total = BSv * TFRM * hw[l] * hw[l] * DM;
    k_flatten<<<(total + 255)/256, 256, 0, stream>>>(F(l*3 + 0), nullptr,           x,   hw[l], hw[l], sOff, total);
    k_flatten<<<(total + 255)/256, 256, 0, stream>>>(F(l*3 + 1), lvl_embed + l*DM,  pos, hw[l], hw[l], sOff, total);
    sOff += hw[l] * hw[l];
  }

  // ---- encoder ----
  for (int i = 0; i < 6; ++i) {
    int B0 = 13 + 16 * i;
    const float *offb = F(B0+1), *awb = F(B0+3), *valb = F(B0+5), *outb = F(B0+7);
    const float *n1g = F(B0+8), *n1b = F(B0+9);
    const float *l1b = F(B0+11), *l2b = F(B0+13);
    const float *n2g = F(B0+14), *n2b = F(B0+15);
    k_add<<<(NETOK*DM + 255)/256, 256, 0, stream>>>(q, x, pos, NETOK*DM);
    gemm(x, e_valw[i], valb, val, NETOK, DM, 0, 256, 256, 0, 0);
    gemm(q, e_offw[i], offb, off, NETOK, DM, 0, 256, 256, 0, 0);
    gemm(q, e_aww[i],  awb,  aw,  NETOK, DM, 0, 128, 128, 0, 0);
    k_softmax<<<(NETOK*HEADS + 255)/256, 256, 0, stream>>>(aw, NETOK*HEADS, 16);
    k_msda_enc<<<(NETOK*HEADS + 255)/256, 256, 0, stream>>>(val, off, aw, attn);
    gemm(attn, e_outw[i], outb, tmp, NETOK, DM, 0, 256, 256, 0, 0);
    k_ln<<<NETOK, 256, 0, stream>>>(x, x, tmp, n1g, n1b);
    gemm(x,   e_l1w[i], l1b, ffn, NETOK, DM,  0, FFN, FFN, 0, 1);   // fused ReLU
    gemm(ffn, e_l2w[i], l2b, tmp, NETOK, FFN, 0, 256, 256, 0, 0);
    k_ln<<<NETOK, 256, 0, stream>>>(x, x, tmp, n2g, n2b);
  }

  // ---- decoder init ----
  k_decinit<<<(NDTOK*DM + 255)/256, 256, 0, stream>>>(F(12), F(242), qpv, xd);
  k_refp<<<(NDTOK + 255)/256, 256, 0, stream>>>(qpv, F(243), F(244), refp);

  // ---- decoder ----
  for (int i = 0; i < 6; ++i) {
    int B0 = 13 + 96 + 22 * i;
    const float *offb = F(B0+1), *awb = F(B0+3), *valb = F(B0+5), *outb = F(B0+7);
    const float *n1g = F(B0+8),  *n1b = F(B0+9);
    const float *binp = F(B0+11);
    const float *woutb = F(B0+13);
    const float *n2g = F(B0+14), *n2b = F(B0+15);
    const float *l1b = F(B0+17), *l2b = F(B0+19);
    const float *n3g = F(B0+20), *n3b = F(B0+21);
    // self attention
    k_add<<<(NDTOK*DM + 255)/256, 256, 0, stream>>>(qd, xd, qpv, NDTOK*DM);
    gemm(qd, d_winw[i], binp, qkv, NDTOK, DM, 0,  512, 768, 0,   0);  // Q,K from (x+qp)
    gemm(xd, d_winw[i], binp, qkv, NDTOK, DM, 32, 256, 768, 512, 0);  // V from x
    k_mha<<<(BSv*HEADS*TALL*NQv + 255)/256, 256, 0, stream>>>(qkv, sa);
    gemm(sa, d_woutw[i], woutb, tmpd, NDTOK, DM, 0, 256, 256, 0, 0);
    k_ln<<<NDTOK, 256, 0, stream>>>(xd, xd, tmpd, n2g, n2b);
    // cross deformable attention over encoder memory (tdim = 4)
    k_add<<<(NDTOK*DM + 255)/256, 256, 0, stream>>>(qd, xd, qpv, NDTOK*DM);
    gemm(x,  d_valw[i], valb, val,  NETOK, DM, 0, 256,  256,  0, 0);
    gemm(qd, d_offw[i], offb, offd, NDTOK, DM, 0, 1024, 1024, 0, 0);
    gemm(qd, d_aww[i],  awb,  awd,  NDTOK, DM, 0, 512,  512,  0, 0);
    k_softmax<<<(NDTOK*HEADS + 255)/256, 256, 0, stream>>>(awd, NDTOK*HEADS, 64);
    k_msda_dec<<<(NDTOK*HEADS + 255)/256, 256, 0, stream>>>(val, offd, awd, refp, attnd);
    gemm(attnd, d_outw[i], outb, tmpd, NDTOK, DM, 0, 256, 256, 0, 0);
    k_ln<<<NDTOK, 256, 0, stream>>>(xd, xd, tmpd, n1g, n1b);
    // FFN
    gemm(xd,   d_l1w[i], l1b, ffnd, NDTOK, DM,  0, FFN, FFN, 0, 1);
    gemm(ffnd, d_l2w[i], l2b, tmpd, NDTOK, FFN, 0, 256, 256, 0, 0);
    k_ln<<<NDTOK, 256, 0, stream>>>(xd, xd, tmpd, n3g, n3b);
  }

  // ---- outputs: hs (2,6,100,256) then refp (2,6,100,2) ----
  float* outp = (float*)d_out;
  k_copy<<<(NDTOK*DM + 255)/256, 256, 0, stream>>>(outp, xd, NDTOK*DM);
  k_copy<<<(NDTOK*2 + 255)/256, 256, 0, stream>>>(outp + NDTOK*DM, refp, NDTOK*2);
}
